// AdditiveMask_69303592288779
// MI455X (gfx1250) — matrix operand order repaired and checked by reading, not compiled, not run on hardware
//
#include <hip/hip_runtime.h>
#include <math.h>

typedef __attribute__((ext_vector_type(16))) _Float16 v16h;
typedef __attribute__((ext_vector_type(8)))  float    v8f;

#define DIM 128
#define EPS 1e-8f

// ---------------------------------------------------------------------------
// Phase 1: h = tanh(x @ W^T + b_lin)  via v_wmma_f32_16x16x32_f16
// One wave computes a 16x16 output tile; block = 256 threads = 8 waves = 8
// column tiles (covers all 128 cols). Grid.x = N/16 (N=50000 -> exact 3125).
// Fragment layouts follow CDNA5 ISA 7.12.2 (wave32):
//   A 16x32 f16 : lane L (g=L>>4, m=L&15) row m; e<8 -> K=8g+e, e>=8 -> K=16+8g+(e-8)
//   B 32x16 f16 : lane L col (L&15); e<8 -> K=8g+e, e>=8 -> K=16+8g+(e-8)  (B[k][n] = W[n][k])
//   C/D 16x16 f32: lane L col (L&15); VGPR r -> row m0 + 8g + r
// ---------------------------------------------------------------------------
__global__ void gemm_tanh_kernel(const float* __restrict__ x,
                                 const float* __restrict__ W,
                                 const float* __restrict__ blin,
                                 float* __restrict__ h, int N)
{
    const int wave = threadIdx.x >> 5;
    const int lane = threadIdx.x & 31;
    const int g    = lane >> 4;
    const int ln   = lane & 15;
    const int m0   = blockIdx.x << 4;
    const int n0   = wave << 4;

    const float* xrow = x + (size_t)(m0 + ln) * DIM;  // A row for this lane
    const float* wrow = W + (size_t)(n0 + ln) * DIM;  // B column (= W row)

    v8f acc = {0.f, 0.f, 0.f, 0.f, 0.f, 0.f, 0.f, 0.f};

    #pragma unroll
    for (int k0 = 0; k0 < DIM; k0 += 32) {
        v16h a, b;
        #pragma unroll
        for (int e = 0; e < 8; ++e)
            a[e] = (_Float16)xrow[k0 + 8 * g + e];
        #pragma unroll
        for (int e = 8; e < 16; ++e)
            a[e] = (_Float16)xrow[k0 + 16 + 8 * g + (e - 8)];
        #pragma unroll
        for (int e = 0; e < 8; ++e)
            b[e] = (_Float16)wrow[k0 + 8 * g + e];
        #pragma unroll
        for (int e = 8; e < 16; ++e)
            b[e] = (_Float16)wrow[k0 + 16 + 8 * g + (e - 8)];
        acc = __builtin_amdgcn_wmma_f32_16x16x32_f16(
                  /*neg_a=*/false, a, /*neg_b=*/false, b,
                  /*c_mod=*/(short)0, acc, /*reuse_a=*/false, /*reuse_b=*/false);
    }

    const float bias = blin[n0 + ln];
    #pragma unroll
    for (int r = 0; r < 8; ++r) {
        int m = m0 + 8 * g + r;
        h[(size_t)m * DIM + n0 + ln] = tanhf(acc[r] + bias);
    }
}

// ---------------------------------------------------------------------------
// Phase 2: hn = h / max(||h||, eps), one wave per row (in place).
// ---------------------------------------------------------------------------
__global__ void normalize_rows_kernel(float* __restrict__ h, int N)
{
    const int row  = blockIdx.x * 8 + (threadIdx.x >> 5);
    const int lane = threadIdx.x & 31;
    if (row >= N) return;

    float4* rp = (float4*)(h + (size_t)row * DIM);
    float4 v = rp[lane];
    float s = v.x * v.x + v.y * v.y + v.z * v.z + v.w * v.w;
    #pragma unroll
    for (int off = 16; off > 0; off >>= 1)
        s += __shfl_xor(s, off, 32);
    const float inv = 1.0f / fmaxf(sqrtf(s), EPS);
    v.x *= inv; v.y *= inv; v.z *= inv; v.w *= inv;
    rp[lane] = v;
}

// ---------------------------------------------------------------------------
// Per-node init: deg = 1 (self loop), h0 = relu(mask), f = h0.
// ---------------------------------------------------------------------------
__global__ void init_node_kernel(const float* __restrict__ mask,
                                 float* __restrict__ deg,
                                 float* __restrict__ h0,
                                 float* __restrict__ f, int N)
{
    int n = blockIdx.x * blockDim.x + threadIdx.x;
    if (n >= N) return;
    deg[n] = 1.0f;
    float m = fmaxf(mask[n], 0.0f);
    h0[n] = m;
    f[n]  = m;
}

// ---------------------------------------------------------------------------
// Phase 3: per-edge cosine (dot of normalized rows), relu, degree accumulate.
// One wave per edge; hn (25.6 MB) is L2-resident on MI455X (192 MB L2).
// ---------------------------------------------------------------------------
__global__ void edge_cos_kernel(const float* __restrict__ hn,
                                const long long* __restrict__ ei,
                                float* __restrict__ ew,
                                float* __restrict__ deg, int E)
{
    const int e    = blockIdx.x * 8 + (threadIdx.x >> 5);
    const int lane = threadIdx.x & 31;
    if (e >= E) return;

    const long long r = ei[e];
    const long long c = ei[(size_t)E + e];
    float4 a = ((const float4*)(hn + (size_t)r * DIM))[lane];
    float4 b = ((const float4*)(hn + (size_t)c * DIM))[lane];
    float s = a.x * b.x + a.y * b.y + a.z * b.z + a.w * b.w;
    #pragma unroll
    for (int off = 16; off > 0; off >>= 1)
        s += __shfl_xor(s, off, 32);

    if (lane == 0) {
        float w = fmaxf(s, 0.0f);
        ew[e] = w;
        atomicAdd(&deg[c], w);
    }
}

// dinv = deg > 0 ? deg^-0.5 : 0   (in place over deg)
__global__ void dinv_kernel(float* __restrict__ deg, int N)
{
    int n = blockIdx.x * blockDim.x + threadIdx.x;
    if (n >= N) return;
    float d = deg[n];
    deg[n] = (d > 0.0f) ? rsqrtf(d) : 0.0f;
}

// enorm[e] = dinv[row] * w[e] * dinv[col]
__global__ void edge_norm_kernel(const long long* __restrict__ ei,
                                 const float* __restrict__ ew,
                                 const float* __restrict__ dinv,
                                 float* __restrict__ enorm, int E)
{
    int e = blockIdx.x * blockDim.x + threadIdx.x;
    if (e >= E) return;
    long long r = ei[e];
    long long c = ei[(size_t)E + e];
    enorm[e] = dinv[r] * ew[e] * dinv[c];
}

// APPNP step a: agg[n] = dinv[n]^2 * f[n]   (self-loop term)
__global__ void self_loop_kernel(const float* __restrict__ dinv,
                                 const float* __restrict__ f,
                                 float* __restrict__ agg, int N)
{
    int n = blockIdx.x * blockDim.x + threadIdx.x;
    if (n >= N) return;
    float di = dinv[n];
    agg[n] = di * di * f[n];
}

// APPNP step b: agg[col] += enorm[e] * f[row]
__global__ void scatter_kernel(const long long* __restrict__ ei,
                               const float* __restrict__ enorm,
                               const float* __restrict__ f,
                               float* __restrict__ agg, int E)
{
    int e = blockIdx.x * blockDim.x + threadIdx.x;
    if (e >= E) return;
    long long r = ei[e];
    long long c = ei[(size_t)E + e];
    atomicAdd(&agg[c], enorm[e] * f[r]);
}

// APPNP step c: f[n] = (1-alpha)*agg[n] + alpha*h0[n]
__global__ void update_kernel(const float* __restrict__ agg,
                              const float* __restrict__ h0,
                              const float* __restrict__ alpha_p,
                              float* __restrict__ f, int N)
{
    int n = blockIdx.x * blockDim.x + threadIdx.x;
    if (n >= N) return;
    float alpha = alpha_p[0];
    f[n] = (1.0f - alpha) * agg[n] + alpha * h0[n];
}

// Phase 5: fill[n] = tanh(f[n] - softplus(bias[0]))
__global__ void fill_kernel(const float* __restrict__ f,
                            const float* __restrict__ bias_p,
                            float* __restrict__ fill, int N)
{
    int n = blockIdx.x * blockDim.x + threadIdx.x;
    if (n >= N) return;
    float b  = bias_p[0];
    float sp = fmaxf(b, 0.0f) + log1pf(expf(-fabsf(b)));  // stable softplus
    fill[n] = tanhf(f[n] - sp);
}

// ---------------------------------------------------------------------------
extern "C" void kernel_launch(void* const* d_in, const int* in_sizes, int n_in,
                              void* d_out, int out_size, void* d_ws, size_t ws_size,
                              hipStream_t stream)
{
    const float*     x     = (const float*)d_in[0];
    const float*     mask  = (const float*)d_in[1];
    const long long* ei    = (const long long*)d_in[2];   // int64 [2, E]
    const float*     W     = (const float*)d_in[3];
    const float*     blin  = (const float*)d_in[4];
    const float*     alpha = (const float*)d_in[5];
    const float*     bias  = (const float*)d_in[6];

    const int N = in_sizes[0] / DIM;    // 50000
    const int E = in_sizes[2] / 2;      // 600000

    float* out_fill = (float*)d_out;            // [N]
    float* out_ew   = (float*)d_out + N;        // [E]

    // Workspace layout (floats):
    float* h     = (float*)d_ws;                // N*128 (becomes hn in place)
    float* deg   = h + (size_t)N * DIM;         // N (becomes dinv in place)
    float* h0    = deg + N;                     // N
    float* f     = h0 + N;                      // N
    float* agg   = f + N;                       // N
    float* enorm = agg + N;                     // E

    const int TB = 256;
    const int gN = (N + TB - 1) / TB;
    const int gE = (E + TB - 1) / TB;

    // 1) h = tanh(x @ W^T + b)  (WMMA)
    gemm_tanh_kernel<<<N / 16, 256, 0, stream>>>(x, W, blin, h, N);
    // 2) normalize rows in place
    normalize_rows_kernel<<<(N + 7) / 8, 256, 0, stream>>>(h, N);
    // 3) node init (deg=1 self loop, h0=relu(mask), f=h0)
    init_node_kernel<<<gN, TB, 0, stream>>>(mask, deg, h0, f, N);
    // 4) edge cosine -> edge_weights(out), deg accumulation
    edge_cos_kernel<<<(E + 7) / 8, 256, 0, stream>>>(h, ei, out_ew, deg, E);
    // 5) dinv in place
    dinv_kernel<<<gN, TB, 0, stream>>>(deg, N);
    // 6) per-edge normalized weights
    edge_norm_kernel<<<gE, TB, 0, stream>>>(ei, out_ew, deg, enorm, E);

    // 7) APPNP, K = 5
    for (int k = 0; k < 5; ++k) {
        self_loop_kernel<<<gN, TB, 0, stream>>>(deg, f, agg, N);
        scatter_kernel<<<gE, TB, 0, stream>>>(ei, enorm, f, agg, E);
        update_kernel<<<gN, TB, 0, stream>>>(agg, h0, alpha, f, N);
    }

    // 8) fill = tanh(f - softplus(bias))
    fill_kernel<<<gN, TB, 0, stream>>>(f, bias, out_fill, N);
}
